// MemoryEfficientAttention_12678743458145
// MI455X (gfx1250) — matrix operand order repaired and checked by reading, not compiled, not run on hardware
//
#include <hip/hip_runtime.h>
#include <hip/hip_bf16.h>

typedef __attribute__((ext_vector_type(16))) _Float16 v16h;
typedef __attribute__((ext_vector_type(8)))  float    v8f;

#define B_  2
#define N_  2048
#define C_  1024
#define H_  16
#define D_  64

// ---------------------------------------------------------------------------
// WMMA helper: D = A(16x32 f16) * B(32x16 f16) + C(16x16 f32)
// ---------------------------------------------------------------------------
__device__ __forceinline__ v8f wmma32(v16h a, v16h b, v8f c) {
  return __builtin_amdgcn_wmma_f32_16x16x32_f16(
      /*neg_a=*/false, a, /*neg_b=*/false, b,
      /*c_mod=*/(short)0, c, /*reuse_a=*/false, /*reuse_b=*/false);
}

// A fragment (16x32 f16): lane m = lane&15.
// elems 0..7  -> K = k0 + 8*(lane>=16) + e ; elems 8..15 -> K = k0+16+8*(lane>=16)+(e-8)
__device__ __forceinline__ v16h load_a_f16(const _Float16* p, int ld, int m0, int k0, int lane) {
  const _Float16* row = p + (size_t)(m0 + (lane & 15)) * ld + k0 + ((lane >> 4) << 3);
  v16h a;
#pragma unroll
  for (int e = 0; e < 8; ++e) a[e] = row[e];
#pragma unroll
  for (int e = 0; e < 8; ++e) a[8 + e] = row[16 + e];
  return a;
}

// B fragment (32x16) with B[k][n] = src[n0+n][k0+k]; row-major src, ld = row stride.
// lane n = lane&15, elems 0..7 -> K = k0 + 8*(lane>=16) + e ; elems 8..15 -> K = k0+16+8*(lane>=16)+(e-8).
__device__ __forceinline__ v16h load_bT_f16(const _Float16* p, int ld, int n0, int k0, int lane) {
  const _Float16* row = p + (size_t)(n0 + (lane & 15)) * ld + k0 + ((lane >> 4) << 3);
  v16h b;
#pragma unroll
  for (int e = 0; e < 8; ++e) b[e] = row[e];
#pragma unroll
  for (int e = 0; e < 8; ++e) b[8 + e] = row[16 + e];
  return b;
}

// 16-lane (half-wave) row reductions for softmax (wave32: xor masks 1..8)
__device__ __forceinline__ float rmax16(float v) {
  v = fmaxf(v, __shfl_xor(v, 1, 32));
  v = fmaxf(v, __shfl_xor(v, 2, 32));
  v = fmaxf(v, __shfl_xor(v, 4, 32));
  v = fmaxf(v, __shfl_xor(v, 8, 32));
  return v;
}
__device__ __forceinline__ float rsum16(float v) {
  v += __shfl_xor(v, 1, 32);
  v += __shfl_xor(v, 2, 32);
  v += __shfl_xor(v, 4, 32);
  v += __shfl_xor(v, 8, 32);
  return v;
}

// ---------------------------------------------------------------------------
// Tensor Data Mover (TDM): stage a 2D f16 tile Global -> LDS.
// D# layout per CDNA5 ISA.
// ---------------------------------------------------------------------------
#if __has_builtin(__builtin_amdgcn_tensor_load_to_lds)
#define HAVE_TDM 1
typedef __attribute__((ext_vector_type(4))) unsigned int u32x4;
typedef __attribute__((ext_vector_type(8))) int i32x8;
typedef __attribute__((ext_vector_type(4))) int i32x4;

__device__ __forceinline__ void tdm_load_2d_f16(
    unsigned int lds_addr, const void* gaddr,
    unsigned int tile_w, unsigned int tile_h,
    unsigned int tdim0, unsigned int tdim1, unsigned int stride0) {
  unsigned long long ga = (unsigned long long)gaddr;
  u32x4 g0;
  g0.x = 1u;                                   // count=1, is_restore=0, no gather
  g0.y = lds_addr;                             // LDS byte address
  g0.z = (unsigned int)ga;                     // global_addr[31:0]
  g0.w = ((unsigned int)(ga >> 32) & 0x01ffffffu) | (2u << 30);  // addr[56:32], type=2
  i32x8 g1;
  g1[0] = 1 << 16;                                             // data_size=1 (2 bytes)
  g1[1] = (int)((tdim0 & 0xffffu) << 16);                      // tensor_dim0[15:0]
  g1[2] = (int)((tdim0 >> 16) | ((tdim1 & 0xffffu) << 16));    // dim0 hi | dim1 lo
  g1[3] = (int)((tdim1 >> 16) | ((tile_w & 0xffffu) << 16));   // dim1 hi | tile_dim0
  g1[4] = (int)(tile_h & 0xffffu);                             // tile_dim1 (tile_dim2=0)
  g1[5] = (int)stride0;                                        // tensor_dim0_stride[31:0]
  g1[6] = 0;                                                   // stride0 hi | stride1 lo
  g1[7] = 0;
  i32x4 z4 = {0, 0, 0, 0};
#if __has_include(<hip/amd_detail/amd_gfx1250_TDM.h>)
  i32x8 z8 = {0, 0, 0, 0, 0, 0, 0, 0};
  __builtin_amdgcn_tensor_load_to_lds(g0, g1, z4, z4, z8, 0);  // clang-23 form
#else
  __builtin_amdgcn_tensor_load_to_lds(g0, g1, z4, z4, 0);      // ROCm 7.2 form
#endif
}
#endif

// ---------------------------------------------------------------------------
// Kernel 0: fp32 -> f16 conversion pre-pass. 8 elems/thread.
// ---------------------------------------------------------------------------
__global__ __launch_bounds__(256) void f32_to_f16_kernel(
    const float* __restrict__ src, _Float16* __restrict__ dst, int count) {
  const int i = (blockIdx.x * 256 + threadIdx.x) * 8;
  if (i + 8 <= count) {
#pragma unroll
    for (int e = 0; e < 8; ++e) dst[i + e] = (_Float16)src[i + e];
  }
}

// ---------------------------------------------------------------------------
// Epilogue helpers
// ---------------------------------------------------------------------------
__device__ __forceinline__ void store_qkv_tile(v8f acc, int m0, int n0, int lane,
                                               const float* __restrict__ bias,
                                               _Float16* __restrict__ qkvh) {
  const int gn = n0 + (lane & 15);
  const float bv = bias[gn];
  const int s = gn >> 10, rm = gn & 1023, h = rm >> 6, d = rm & 63;
#pragma unroll
  for (int r = 0; r < 8; ++r) {
    const int gm = m0 + r + ((lane >> 4) << 3);
    const int bb = gm >> 11, nn = gm & 2047;
    qkvh[((((size_t)s * B_ + bb) * H_ + h) * N_ + nn) * D_ + d] = (_Float16)(acc[r] + bv);
  }
}

__device__ __forceinline__ void store_out_tile(v8f acc, int m0, int n0, int lane,
                                               const float* __restrict__ bias,
                                               float* __restrict__ out) {
  const int gn = n0 + (lane & 15);
  const float bv = bias[gn];
#pragma unroll
  for (int r = 0; r < 8; ++r) {
    const int gm = m0 + r + ((lane >> 4) << 3);
    out[(size_t)gm * C_ + gn] = acc[r] + bv;
  }
}

// ---------------------------------------------------------------------------
// Kernel 1: QKV GEMM (f16 in, f32 accum).  M=4096, Nout=3072, K=1024.
// 8 waves/block, wave = 32x32 output (2x2 tiles, 4 WMMA / K-step).
// ---------------------------------------------------------------------------
__global__ __launch_bounds__(256) void qkv_gemm_kernel(
    const _Float16* __restrict__ xh, const _Float16* __restrict__ wh,
    const float* __restrict__ bias, _Float16* __restrict__ qkvh) {
  const int lane = threadIdx.x & 31;
  const int wave = threadIdx.x >> 5;
  const int bm = blockIdx.x & 63;
  const int bn = blockIdx.x >> 6;
  const int m0 = bm * 64 + (wave >> 2) * 32;
  const int n0 = bn * 128 + (wave & 3) * 32;

  v8f acc00 = {}, acc01 = {}, acc10 = {}, acc11 = {};
  const _Float16* arow = xh + (size_t)(m0 + (lane & 15)) * C_;
  const _Float16* brow = wh + (size_t)(n0 + (lane & 15)) * C_;
  for (int k0 = 0; k0 < C_; k0 += 32) {
    __builtin_prefetch(arow + k0 + 32, 0, 3);   // global_prefetch_b8
    __builtin_prefetch(brow + k0 + 32, 0, 3);
    v16h a0 = load_a_f16(xh, C_, m0, k0, lane);
    v16h a1 = load_a_f16(xh, C_, m0 + 16, k0, lane);
    v16h b0 = load_bT_f16(wh, C_, n0, k0, lane);
    v16h b1 = load_bT_f16(wh, C_, n0 + 16, k0, lane);
    acc00 = wmma32(a0, b0, acc00);
    acc01 = wmma32(a0, b1, acc01);
    acc10 = wmma32(a1, b0, acc10);
    acc11 = wmma32(a1, b1, acc11);
  }
  store_qkv_tile(acc00, m0, n0, lane, bias, qkvh);
  store_qkv_tile(acc01, m0, n0 + 16, lane, bias, qkvh);
  store_qkv_tile(acc10, m0 + 16, n0, lane, bias, qkvh);
  store_qkv_tile(acc11, m0 + 16, n0 + 16, lane, bias, qkvh);
}

// ---------------------------------------------------------------------------
// Kernel 2: RoPE + reorg. Rotates q (x 1/64 = D^-.5 * D^-.5) and k, writes
// q/k as [b][h][n][d] f16 and v transposed as [b][h][d][n] f16.
// ---------------------------------------------------------------------------
__global__ __launch_bounds__(256) void rope_reorg_kernel(
    const _Float16* __restrict__ qkvh, _Float16* __restrict__ qr,
    _Float16* __restrict__ kr, _Float16* __restrict__ vt) {
  const int tid = blockIdx.x * blockDim.x + threadIdx.x;
  const int j = tid & 31;
  const int n = (tid >> 5) & (N_ - 1);
  const int h = (tid >> 16) & (H_ - 1);
  const int b = tid >> 20;

  const size_t S = (size_t)B_ * H_ * N_ * D_;
  const size_t base = (((size_t)b * H_ + h) * N_ + n) * D_;

  const float inv = __expf(-0.2878231366242557f * (float)j);  // 10000^(-j/32)
  const float ang = (float)n * inv;
  const float c = cosf(ang), si = sinf(ang);

  const float qe = (float)qkvh[base + 2 * j];
  const float qo = (float)qkvh[base + 2 * j + 1];
  qr[base + j]      = (_Float16)((qe * c - qo * si) * 0.015625f);
  qr[base + j + 32] = (_Float16)((qo * c + qe * si) * 0.015625f);

  const float ke = (float)qkvh[S + base + 2 * j];
  const float ko = (float)qkvh[S + base + 2 * j + 1];
  kr[base + j]      = (_Float16)(ke * c - ko * si);
  kr[base + j + 32] = (_Float16)(ko * c + ke * si);

  const size_t vtb = ((size_t)b * H_ + h) * D_;
  vt[(vtb + j) * N_ + n]      = qkvh[2 * S + base + j];
  vt[(vtb + j + 32) * N_ + n] = qkvh[2 * S + base + j + 32];
}

// ---------------------------------------------------------------------------
// Kernel 3: flash attention. grid = (B*H)*(N/128) = 512 blocks, 256 thr = 8 waves.
// Double-buffered TDM: wave0 issues next tile's DMA into the ping-pong buffer,
// waits tensorcnt<=2 (TDM completes in order) so the current tile is ready
// while the next streams in.  8 waves x 16 q-rows share each staged tile.
// ---------------------------------------------------------------------------
__global__ __launch_bounds__(256) void flash_attn_kernel(
    const _Float16* __restrict__ qr, const _Float16* __restrict__ kr,
    const _Float16* __restrict__ vt, float* __restrict__ attn) {
  __shared__ _Float16 ktile[2][32][64];   // [buf][k-row][d]
  __shared__ _Float16 vtile[2][64][32];   // [buf][d][k-row]
  __shared__ _Float16 plds[8][16][32];    // per-wave P relayout scratch

  const int lane = threadIdx.x & 31;
  const int wave = threadIdx.x >> 5;
  const int bh = blockIdx.x >> 4;         // 0..31  (b*16 + h)
  const int qt = blockIdx.x & 15;         // q tile of 128 rows
  const int b = bh >> 4, h = bh & 15;
  const int q0 = qt * 128 + wave * 16;

  const _Float16* qbh = qr + (size_t)bh * N_ * D_;
  const _Float16* kbh = kr + (size_t)bh * N_ * D_;
  const _Float16* vbh = vt + (size_t)bh * D_ * N_;   // [d][n]

  const v16h aq0 = load_a_f16(qbh, D_, q0, 0, lane);
  const v16h aq1 = load_a_f16(qbh, D_, q0, 32, lane);

#ifdef HAVE_TDM
  const unsigned int klds0 = (unsigned int)(size_t)(&ktile[0][0][0]);
  const unsigned int klds1 = (unsigned int)(size_t)(&ktile[1][0][0]);
  const unsigned int vlds0 = (unsigned int)(size_t)(&vtile[0][0][0]);
  const unsigned int vlds1 = (unsigned int)(size_t)(&vtile[1][0][0]);
  if (wave == 0) {   // prologue: stage tile 0 into buffer 0
    tdm_load_2d_f16(klds0, kbh, D_, 32, D_, N_, D_);
    tdm_load_2d_f16(vlds0, vbh, 32, D_, N_, D_, N_);
  }
#endif

  float mrun[8], lrun[8];
  v8f o0 = {}, o1 = {}, o2 = {}, o3 = {};
#pragma unroll
  for (int r = 0; r < 8; ++r) { mrun[r] = -1e30f; lrun[r] = 0.0f; }

  for (int kb = 0; kb < N_; kb += 32) {
#ifdef HAVE_TDM
    const int buf = (kb >> 5) & 1;
    if (wave == 0) {
      if (kb + 32 < N_) {
        tdm_load_2d_f16(buf ? klds0 : klds1, kbh + (size_t)(kb + 32) * D_,
                        D_, 32, D_, N_, D_);
        tdm_load_2d_f16(buf ? vlds0 : vlds1, vbh + (kb + 32),
                        32, D_, N_, D_, N_);
        __builtin_amdgcn_s_wait_tensorcnt(2);   // current tile's loads done
      } else {
        __builtin_amdgcn_s_wait_tensorcnt(0);
      }
    }
    __syncthreads();                            // tile visible to all waves
    const _Float16* kt = &ktile[buf][0][0];
    const _Float16* vp = &vtile[buf][0][0];
    v8f c0 = {}, c1 = {};
    c0 = wmma32(aq0, load_bT_f16(kt, 64, 0,  0,  lane), c0);
    c0 = wmma32(aq1, load_bT_f16(kt, 64, 0,  32, lane), c0);
    c1 = wmma32(aq0, load_bT_f16(kt, 64, 16, 0,  lane), c1);
    c1 = wmma32(aq1, load_bT_f16(kt, 64, 16, 32, lane), c1);
#else
    v8f c0 = {}, c1 = {};
    c0 = wmma32(aq0, load_bT_f16(kbh, D_, kb,      0,  lane), c0);
    c0 = wmma32(aq1, load_bT_f16(kbh, D_, kb,      32, lane), c0);
    c1 = wmma32(aq0, load_bT_f16(kbh, D_, kb + 16, 0,  lane), c1);
    c1 = wmma32(aq1, load_bT_f16(kbh, D_, kb + 16, 32, lane), c1);
#endif

    float p0[8], p1[8];
#pragma unroll
    for (int r = 0; r < 8; ++r) {
      const float mx = fmaxf(rmax16(c0[r]), rmax16(c1[r]));
      const float mn = fmaxf(mrun[r], mx);
      const float sc = __expf(mrun[r] - mn);
      mrun[r] = mn;
      p0[r] = __expf(c0[r] - mn);
      p1[r] = __expf(c1[r] - mn);
      lrun[r] = lrun[r] * sc + rsum16(p0[r] + p1[r]);
      o0[r] *= sc; o1[r] *= sc; o2[r] *= sc; o3[r] *= sc;
    }

    // C-layout (row = r + 8*(lane>=16), col = lane&15) -> LDS -> A-layout
    const int mbase = (lane >> 4) << 3;
    const int nn = lane & 15;
#pragma unroll
    for (int r = 0; r < 8; ++r) {
      plds[wave][mbase + r][nn]      = (_Float16)p0[r];
      plds[wave][mbase + r][16 + nn] = (_Float16)p1[r];
    }
    v16h pa;
    {
      const int m = lane & 15;
      const int koff = (lane >> 4) << 3;
#pragma unroll
      for (int e = 0; e < 8; ++e) pa[e]     = plds[wave][m][koff + e];
#pragma unroll
      for (int e = 0; e < 8; ++e) pa[8 + e] = plds[wave][m][16 + koff + e];
    }

#ifdef HAVE_TDM
    o0 = wmma32(pa, load_bT_f16(vp, 32, 0,  0, lane), o0);
    o1 = wmma32(pa, load_bT_f16(vp, 32, 16, 0, lane), o1);
    o2 = wmma32(pa, load_bT_f16(vp, 32, 32, 0, lane), o2);
    o3 = wmma32(pa, load_bT_f16(vp, 32, 48, 0, lane), o3);
    __syncthreads();     // all waves done with buf before it is re-filled
#else
    o0 = wmma32(pa, load_bT_f16(vbh, N_, 0,  kb, lane), o0);
    o1 = wmma32(pa, load_bT_f16(vbh, N_, 16, kb, lane), o1);
    o2 = wmma32(pa, load_bT_f16(vbh, N_, 32, kb, lane), o2);
    o3 = wmma32(pa, load_bT_f16(vbh, N_, 48, kb, lane), o3);
#endif
  }

  const int nn = lane & 15;
#pragma unroll
  for (int r = 0; r < 8; ++r) {
    const int m = q0 + r + ((lane >> 4) << 3);
    const float invl = 1.0f / lrun[r];
    const size_t row = ((size_t)b * N_ + m) * C_ + (size_t)h * D_;
    attn[row + nn]      = o0[r] * invl;
    attn[row + 16 + nn] = o1[r] * invl;
    attn[row + 32 + nn] = o2[r] * invl;
    attn[row + 48 + nn] = o3[r] * invl;
  }
}

// ---------------------------------------------------------------------------
// Kernel 4: output projection (f16 in). M=4096, Nout=1024, K=1024.
// ---------------------------------------------------------------------------
__global__ __launch_bounds__(256) void proj_gemm_kernel(
    const _Float16* __restrict__ ah, const _Float16* __restrict__ wh,
    const float* __restrict__ bias, float* __restrict__ out) {
  const int lane = threadIdx.x & 31;
  const int wave = threadIdx.x >> 5;
  const int bm = blockIdx.x & 63;
  const int bn = blockIdx.x >> 6;
  const int m0 = bm * 64 + (wave >> 2) * 32;
  const int n0 = bn * 128 + (wave & 3) * 32;

  v8f acc00 = {}, acc01 = {}, acc10 = {}, acc11 = {};
  for (int k0 = 0; k0 < C_; k0 += 32) {
    v16h a0 = load_a_f16(ah, C_, m0, k0, lane);
    v16h a1 = load_a_f16(ah, C_, m0 + 16, k0, lane);
    v16h b0 = load_bT_f16(wh, C_, n0, k0, lane);
    v16h b1 = load_bT_f16(wh, C_, n0 + 16, k0, lane);
    acc00 = wmma32(a0, b0, acc00);
    acc01 = wmma32(a0, b1, acc01);
    acc10 = wmma32(a1, b0, acc10);
    acc11 = wmma32(a1, b1, acc11);
  }
  store_out_tile(acc00, m0, n0, lane, bias, out);
  store_out_tile(acc01, m0, n0 + 16, lane, bias, out);
  store_out_tile(acc10, m0 + 16, n0, lane, bias, out);
  store_out_tile(acc11, m0 + 16, n0 + 16, lane, bias, out);
}

// ---------------------------------------------------------------------------
// Workspace layout (88 MiB used):
//   [0,        +8388608)   xh    f16 [4096][1024]
//   [8388608,  +6291456)   wqh   f16 [3072][1024]
//   [14680064, +2097152)   wph   f16 [1024][1024]
//   [16777216, +25165824)  qkvh  f16 [3][B][H][N][D]
//   [41943040, +8388608)   qr    f16 [B][H][N][D]
//   [50331648, +8388608)   kr    f16 [B][H][N][D]
//   [58720256, +8388608)   vt    f16 [B][H][D][N]
//   [67108864, +16777216)  attn  f32 [B][N][C]
//   [83886080, +8388608)   attnh f16 [B][N][C]
// ---------------------------------------------------------------------------
extern "C" void kernel_launch(void* const* d_in, const int* in_sizes, int n_in,
                              void* d_out, int out_size, void* d_ws, size_t ws_size,
                              hipStream_t stream) {
  const float* x      = (const float*)d_in[0];
  const float* qkv_w  = (const float*)d_in[1];
  const float* qkv_b  = (const float*)d_in[2];
  const float* proj_w = (const float*)d_in[3];
  const float* proj_b = (const float*)d_in[4];
  float* out = (float*)d_out;

  char* ws = (char*)d_ws;
  _Float16* xh    = (_Float16*)(ws);
  _Float16* wqh   = (_Float16*)(ws + 8388608);
  _Float16* wph   = (_Float16*)(ws + 14680064);
  _Float16* qkvh  = (_Float16*)(ws + 16777216);
  _Float16* qr    = (_Float16*)(ws + 41943040);
  _Float16* kr    = (_Float16*)(ws + 50331648);
  _Float16* vt    = (_Float16*)(ws + 58720256);
  float*    attn  = (float*)   (ws + 67108864);
  _Float16* attnh = (_Float16*)(ws + 83886080);

  const int nx  = B_ * N_ * C_;       // 4194304
  const int nwq = 3 * C_ * C_;        // 3145728
  const int nwp = C_ * C_;            // 1048576

  f32_to_f16_kernel<<<nx / 2048, 256, 0, stream>>>(x, xh, nx);
  f32_to_f16_kernel<<<nwq / 2048, 256, 0, stream>>>(qkv_w, wqh, nwq);
  f32_to_f16_kernel<<<nwp / 2048, 256, 0, stream>>>(proj_w, wph, nwp);

  qkv_gemm_kernel<<<64 * 24, 256, 0, stream>>>(xh, wqh, qkv_b, qkvh);
  rope_reorg_kernel<<<(B_ * H_ * N_ * 32) / 256, 256, 0, stream>>>(qkvh, qr, kr, vt);
  flash_attn_kernel<<<B_ * H_ * (N_ / 128), 256, 0, stream>>>(qr, kr, vt, attn);
  f32_to_f16_kernel<<<nx / 2048, 256, 0, stream>>>(attn, attnh, nx);
  proj_gemm_kernel<<<64 * 8, 256, 0, stream>>>(attnh, wph, proj_b, out);
}
